// SelfAttention_53317724013112
// MI455X (gfx1250) — matrix operand order repaired and checked by reading, not compiled, not run on hardware
//
#include <hip/hip_runtime.h>
#include <hip/hip_fp16.h>
#include <cstdint>

// MI455X / gfx1250, wave32. WMMA f32_16x16x32_f16 for all matmuls,
// GLOBAL_LOAD_ASYNC_TO_LDS_B128 double-buffered staging in flash attention.
// Self-attention: b=4, n=2048, c=512, h=8, hd=64.

typedef __attribute__((ext_vector_type(16))) _Float16 v16h;
typedef __attribute__((ext_vector_type(8)))  float    v8f;

#define DIM   512
#define HEADS 8
#define HD    64
#define SEQ   2048
#define BATCH 4

#define LDS32 40   // LDS row stride (halfs) for 32-col tiles: 80B rows (16B aligned)
#define LDS64 72   // LDS row stride (halfs) for 64-col tiles: 144B rows

// Async copy: 16B per lane, memory -> LDS, tracked by ASYNCcnt (CDNA5 path).
__device__ __forceinline__ void async_b128(uint32_t lds_off, uint32_t voff,
                                           const _Float16* sbase) {
    asm volatile("global_load_async_to_lds_b128 %0, %1, %2"
                 :: "v"(lds_off), "v"(voff), "s"(sbase)
                 : "memory");
}
__device__ __forceinline__ void wait_async0() {
    asm volatile("s_wait_asynccnt 0x0" ::: "memory");
}

// ---------------------------------------------------------------------------
// Tiled GEMM: Y = A[8192x512] @ W[N x 512]^T + bias, staged fp32->f16 in LDS.
// mode 1: scatter f16 into qkvh: Q,K as [b][h][n][hd]; V transposed [b][h][hd][n]
// mode 0: fp32 row-major out [8192 x 512] (output projection, N=512)
// Block: 128 threads (4 waves), tile 128x128; each wave owns 64x64.
// ---------------------------------------------------------------------------
__global__ __launch_bounds__(128)
void gemm_qkv_proj(const float* __restrict__ A,
                   const float* __restrict__ W,
                   const float* __restrict__ bias,
                   float* __restrict__ outf,
                   _Float16* __restrict__ qkvh,
                   int mode)
{
    __shared__ _Float16 As[128 * LDS32];
    __shared__ _Float16 Bs[128 * LDS32];

    const int t      = threadIdx.x;
    const int lane   = t & 31;
    const int w      = t >> 5;
    const int l15    = lane & 15;
    const int half16 = lane >> 4;
    const int wm     = (w & 1) * 64;
    const int wn     = (w >> 1) * 64;
    const int m0     = blockIdx.x * 128;
    const int n0     = blockIdx.y * 128;

    v8f acc[4][4];
#pragma unroll
    for (int i = 0; i < 4; i++)
#pragma unroll
        for (int j = 0; j < 4; j++)
#pragma unroll
            for (int e = 0; e < 8; e++) acc[i][j][e] = 0.0f;

    for (int k0 = 0; k0 < DIM; k0 += 32) {
        __syncthreads();
#pragma unroll
        for (int i = 0; i < 8; i++) {
            int idx = t + i * 128;        // float4 index within the 128x32 tile
            int row = idx >> 3;
            int c4  = idx & 7;
            float4 av = *reinterpret_cast<const float4*>(A + (size_t)(m0 + row) * DIM + k0 + c4 * 4);
            _Float16* d = &As[row * LDS32 + c4 * 4];
            d[0] = (_Float16)av.x; d[1] = (_Float16)av.y;
            d[2] = (_Float16)av.z; d[3] = (_Float16)av.w;
            float4 bv = *reinterpret_cast<const float4*>(W + (size_t)(n0 + row) * DIM + k0 + c4 * 4);
            _Float16* e = &Bs[row * LDS32 + c4 * 4];
            e[0] = (_Float16)bv.x; e[1] = (_Float16)bv.y;
            e[2] = (_Float16)bv.z; e[3] = (_Float16)bv.w;
        }
        __syncthreads();

        // A-frag: lanes0-15 row=M, K=0..15 contiguous; lanes16-31 K=16..31
        v16h a[4], b[4];
#pragma unroll
        for (int i = 0; i < 4; i++) {
            const _Float16* p = &As[(wm + 16 * i + l15) * LDS32 + half16 * 16];
#pragma unroll
            for (int q = 0; q < 16; q++) a[i][q] = p[q];
        }
        // B-frag: lanes0-15 N=col, K=0..15 contiguous; lanes16-31 K=16..31
#pragma unroll
        for (int j = 0; j < 4; j++) {
            const _Float16* p = &Bs[(wn + 16 * j + l15) * LDS32 + half16 * 16];
#pragma unroll
            for (int q = 0; q < 16; q++) b[j][q] = p[q];
        }
#pragma unroll
        for (int i = 0; i < 4; i++)
#pragma unroll
            for (int j = 0; j < 4; j++)
                acc[i][j] = __builtin_amdgcn_wmma_f32_16x16x32_f16(
                    false, a[i], false, b[j], (short)0, acc[i][j], false, false);
    }

    // Epilogue: C layout -> row = 8*half16 + vgpr, col = l15 (per 16x16 tile)
#pragma unroll
    for (int i = 0; i < 4; i++) {
#pragma unroll
        for (int j = 0; j < 4; j++) {
            int gcol = n0 + wn + 16 * j + l15;
            float bv = bias[gcol];
#pragma unroll
            for (int r = 0; r < 8; r++) {
                int grow = m0 + wm + 16 * i + 8 * half16 + r;
                float v  = acc[i][j][r] + bv;
                if (mode == 0) {
                    outf[(size_t)grow * DIM + gcol] = v;
                } else {
                    int which = gcol >> 9;     // 0=q 1=k 2=v
                    int c     = gcol & 511;
                    int hh    = c >> 6;
                    int hd    = c & 63;
                    int bb    = grow >> 11;    // /2048
                    int nn    = grow & 2047;
                    size_t off;
                    if (which == 2)            // V transposed per head: [b][h][hd][n]
                        off = (((size_t)(2 * BATCH + bb) * HEADS + hh) * HD + hd) * SEQ + nn;
                    else                       // Q,K: [b][h][n][hd]
                        off = ((((size_t)which * BATCH + bb) * HEADS + hh) * SEQ + nn) * HD + hd;
                    qkvh[off] = (_Float16)v;
                }
            }
        }
    }
}

// ---------------------------------------------------------------------------
// Flash attention: one block (128 threads, 4 waves) per (b,h, 64-row q tile).
// Q/K/V tiles staged via async-to-LDS b128 copies; K/V double-buffered so the
// copy of tile t+1 overlaps the WMMA + softmax work of tile t.
// ---------------------------------------------------------------------------
__global__ __launch_bounds__(128)
void flash_attn(const _Float16* __restrict__ qkvh, float* __restrict__ attnout)
{
    __shared__ _Float16 Qs[64 * LDS64];
    __shared__ _Float16 Ks[2][64 * LDS64];
    __shared__ _Float16 Vts[2][64 * LDS64];
    __shared__ _Float16 Ss[64 * LDS64];

    const int t      = threadIdx.x;
    const int lane   = t & 31;
    const int w      = t >> 5;
    const int l15    = lane & 15;
    const int half16 = lane >> 4;
    const int qt     = blockIdx.x;
    const int bh     = blockIdx.y;
    const int b      = bh >> 3;
    const int h      = bh & 7;

    const size_t plane = (size_t)SEQ * HD;
    const _Float16* Qg = qkvh + ((size_t)(0 * BATCH + b) * HEADS + h) * plane;
    const _Float16* Kg = qkvh + ((size_t)(1 * BATCH + b) * HEADS + h) * plane;
    const _Float16* Vg = qkvh + ((size_t)(2 * BATCH + b) * HEADS + h) * plane; // [hd][n]

    const uint32_t ldsQ  = (uint32_t)(uintptr_t)&Qs[0];
    const uint32_t ldsK0 = (uint32_t)(uintptr_t)&Ks[0][0];
    const uint32_t ldsV0 = (uint32_t)(uintptr_t)&Vts[0][0];
    const uint32_t tileb = 64 * LDS64 * 2;      // bytes per buffer

    // Per-wave async issue: 4 instructions x 32 lanes x 16B = 2KB; 4 waves = 8KB tile.
    auto issue_kv = [&](int kt2, int buf) {
        uint32_t ldsK = ldsK0 + (uint32_t)buf * tileb;
        uint32_t ldsV = ldsV0 + (uint32_t)buf * tileb;
#pragma unroll
        for (int i = 0; i < 4; i++) {
            int c   = (w * 4 + i) * 32 + lane;  // 16B chunk id, 512 per tile
            int row = c >> 3;
            int c8  = c & 7;
            uint32_t dlo = (uint32_t)(row * (LDS64 * 2) + c8 * 16);
            // K tile: rows n (contiguous 8KB region)
            async_b128(ldsK + dlo, (uint32_t)(kt2 * 8192 + c * 16), Kg);
            // V tile: rows hd (row stride SEQ halfs), cols = key window
            async_b128(ldsV + dlo, (uint32_t)(row * (SEQ * 2) + kt2 * 128 + c8 * 16), Vg);
        }
    };

    // Prologue: Q tile + first K/V tile
#pragma unroll
    for (int i = 0; i < 4; i++) {
        int c   = (w * 4 + i) * 32 + lane;
        int row = c >> 3;
        int c8  = c & 7;
        async_b128(ldsQ + (uint32_t)(row * (LDS64 * 2) + c8 * 16),
                   (uint32_t)(qt * 8192 + c * 16), Qg);
    }
    issue_kv(0, 0);
    wait_async0();
    __syncthreads();

    // Q A-frags for the wave's 16 rows (contraction over hd, two 32-chunks)
    v16h qa[2];
#pragma unroll
    for (int d = 0; d < 2; d++) {
        const _Float16* p = &Qs[(16 * w + l15) * LDS64 + 32 * d + half16 * 16];
#pragma unroll
        for (int q = 0; q < 16; q++) qa[d][q] = p[q];
    }

    float mrow[8], lrow[8];
    v8f oacc[4];
#pragma unroll
    for (int r = 0; r < 8; r++) { mrow[r] = -1e30f; lrow[r] = 0.0f; }
#pragma unroll
    for (int j = 0; j < 4; j++)
#pragma unroll
        for (int e = 0; e < 8; e++) oacc[j][e] = 0.0f;

    const float scale = 0.125f;     // 1/sqrt(64)

    for (int kt = 0; kt < SEQ / 64; kt++) {
        // Kick off next tile's async copies; they overlap this tile's compute.
        if (kt + 1 < SEQ / 64) issue_kv(kt + 1, (kt + 1) & 1);
        const int cur = kt & 1;
        const _Float16* ksp = Ks[cur];
        const _Float16* vtp = Vts[cur];

        // S[16 x 64] = Q_tile @ K_tile^T
        v8f sacc[4];
#pragma unroll
        for (int j = 0; j < 4; j++)
#pragma unroll
            for (int e = 0; e < 8; e++) sacc[j][e] = 0.0f;
#pragma unroll
        for (int j = 0; j < 4; j++) {
#pragma unroll
            for (int d = 0; d < 2; d++) {
                v16h kb;
                const _Float16* p = &ksp[(16 * j + l15) * LDS64 + 32 * d + half16 * 16];
#pragma unroll
                for (int q = 0; q < 16; q++) kb[q] = p[q];
                sacc[j] = __builtin_amdgcn_wmma_f32_16x16x32_f16(
                    false, qa[d], false, kb, (short)0, sacc[j], false, false);
            }
        }

        // Online softmax; row r lives in vgpr slot r across the 16-lane half.
        float corr[8];
#pragma unroll
        for (int r = 0; r < 8; r++) {
            float mx = fmaxf(fmaxf(sacc[0][r], sacc[1][r]), fmaxf(sacc[2][r], sacc[3][r])) * scale;
#pragma unroll
            for (int msk = 1; msk < 16; msk <<= 1)
                mx = fmaxf(mx, __shfl_xor(mx, msk, 32));
            float mnew = fmaxf(mrow[r], mx);
            corr[r] = __expf(mrow[r] - mnew);
            mrow[r] = mnew;
        }
        float rsum[8];
#pragma unroll
        for (int r = 0; r < 8; r++) rsum[r] = 0.0f;
#pragma unroll
        for (int j = 0; j < 4; j++) {
#pragma unroll
            for (int r = 0; r < 8; r++) {
                float p = __expf(sacc[j][r] * scale - mrow[r]);
                rsum[r] += p;
                Ss[(16 * w + 8 * half16 + r) * LDS64 + 16 * j + l15] = (_Float16)p;
            }
        }
#pragma unroll
        for (int r = 0; r < 8; r++) {
            float s = rsum[r];
#pragma unroll
            for (int msk = 1; msk < 16; msk <<= 1)
                s += __shfl_xor(s, msk, 32);
            lrow[r] = lrow[r] * corr[r] + s;
#pragma unroll
            for (int j = 0; j < 4; j++) oacc[j][r] *= corr[r];
        }

        // O[16 x 64] += P @ V_tile (P read back as A-frags; wave-local LDS RAW)
#pragma unroll
        for (int k2 = 0; k2 < 2; k2++) {
            v16h pa;
            const _Float16* pp = &Ss[(16 * w + l15) * LDS64 + 32 * k2 + half16 * 16];
#pragma unroll
            for (int q = 0; q < 16; q++) pa[q] = pp[q];
#pragma unroll
            for (int j = 0; j < 4; j++) {
                v16h vb;
                const _Float16* vp = &vtp[(16 * j + l15) * LDS64 + 32 * k2 + half16 * 16];
#pragma unroll
                for (int q = 0; q < 16; q++) vb[q] = vp[q];
                oacc[j] = __builtin_amdgcn_wmma_f32_16x16x32_f16(
                    false, pa, false, vb, (short)0, oacc[j], false, false);
            }
        }

        // Drain next tile's copies and separate this tile's reads from the
        // following iteration's buffer overwrites.
        wait_async0();
        __syncthreads();
    }

    // Normalize and write [b][n][c] fp32 for the projection GEMM
#pragma unroll
    for (int r = 0; r < 8; r++) {
        float inv = 1.0f / lrow[r];
        int grow = qt * 64 + 16 * w + 8 * half16 + r;
#pragma unroll
        for (int j = 0; j < 4; j++) {
            int gcol = h * 64 + 16 * j + l15;
            attnout[((size_t)b * SEQ + grow) * DIM + gcol] = oacc[j][r] * inv;
        }
    }
}

extern "C" void kernel_launch(void* const* d_in, const int* in_sizes, int n_in,
                              void* d_out, int out_size, void* d_ws, size_t ws_size,
                              hipStream_t stream) {
    const float* x      = (const float*)d_in[0];
    const float* qkv_w  = (const float*)d_in[1];
    const float* qkv_b  = (const float*)d_in[2];
    const float* proj_w = (const float*)d_in[3];
    const float* proj_b = (const float*)d_in[4];
    float* out = (float*)d_out;

    // Workspace: f16 QKV (25.2 MB) + f32 attn-out [b][n][c] (16.8 MB)
    _Float16* qkvh = (_Float16*)d_ws;
    float* attnout = (float*)((char*)d_ws + (size_t)3 * BATCH * HEADS * SEQ * HD * sizeof(_Float16));

    // 1) QKV projection: [8192 x 512] @ [1536 x 512]^T
    gemm_qkv_proj<<<dim3(BATCH * SEQ / 128, 3 * DIM / 128), 128, 0, stream>>>(
        x, qkv_w, qkv_b, nullptr, qkvh, 1);
    // 2) Flash attention per (b,h,qtile)
    flash_attn<<<dim3(SEQ / 64, BATCH * HEADS), 128, 0, stream>>>(qkvh, attnout);
    // 3) Output projection: [8192 x 512] @ [512 x 512]^T
    gemm_qkv_proj<<<dim3(BATCH * SEQ / 128, DIM / 128), 128, 0, stream>>>(
        attnout, proj_w, proj_b, out, nullptr, 0);
}
